// MultiHorizonTemporalTransformer_21775484191194
// MI455X (gfx1250) — matrix operand order repaired and checked by reading, not compiled, not run on hardware
//
#include <hip/hip_runtime.h>
#include <hip/hip_bf16.h>

// ---------------- problem constants ----------------
static constexpr int Bn = 4, Sn = 1024, Fn = 20, Dn = 512, Hn = 8, Ln = 4;
static constexpr int FFn = 2048, Mrel = 32, NHn = 3, D2n = 256;
static constexpr int HD = Dn / Hn;           // 64
static constexpr int RELN = 2 * Mrel + 1;    // 65
static constexpr int NROW = Bn * Sn;         // 4096

typedef __attribute__((ext_vector_type(16))) _Float16 v16h;
typedef __attribute__((ext_vector_type(8)))  _Float16 v8h;
typedef __attribute__((ext_vector_type(4)))  _Float16 v4h;
typedef __attribute__((ext_vector_type(8)))  float    v8f;

union AFrag { v16h v; v8h h[2]; _Float16 e[16]; };
union CFrag { v8f  v; float    e[8];  };

#define WMMA_F16(A, Bm, C) __builtin_amdgcn_wmma_f32_16x16x32_f16(false, (A), false, (Bm), (short)0, (C), false, false)

// ---- fragment builders (CDNA5 16x16x32 f16 layouts, wave32) ----
__device__ __forceinline__ v16h load_row_frag(const _Float16* base, int ld, int lane) {
    int m = lane & 15, kh = (lane >> 4) & 1;
    AFrag a;
    a.h[0] = *(const v8h*)(base + m * ld + kh * 8);
    a.h[1] = *(const v8h*)(base + m * ld + 16 + kh * 8);
    return a.v;
}
__device__ __forceinline__ v16h load_colT_frag(const _Float16* baseT, int ld, int lane) {
    int n = lane & 15, kh = (lane >> 4) & 1;
    AFrag b;
    b.h[0] = *(const v8h*)(baseT + n * ld + kh * 8);
    b.h[1] = *(const v8h*)(baseT + n * ld + 16 + kh * 8);
    return b.v;
}

__device__ __forceinline__ v4h cvt4(float4 f) {
    v4h h;
    h[0] = (_Float16)f.x; h[1] = (_Float16)f.y;
    h[2] = (_Float16)f.z; h[3] = (_Float16)f.w;
    return h;
}

__device__ __forceinline__ float gelu_f(float x) {
    return 0.5f * x * (1.0f + erff(x * 0.70710678118654752f));
}

// ---------------- input projection + LN + posenc ----------------
__global__ __launch_bounds__(256) void input_proj_k(const float* __restrict__ x,
                                                    const float* __restrict__ W,
                                                    const float* __restrict__ bias,
                                                    const float* __restrict__ g,
                                                    const float* __restrict__ be,
                                                    float* __restrict__ h) {
    __shared__ float xr[Fn];
    __shared__ float rowv[Dn];
    __shared__ float red[256];
    int row = blockIdx.x, tid = threadIdx.x;
    int s = row % Sn;
    if (tid < Fn) xr[tid] = x[(size_t)row * Fn + tid];
    __syncthreads();
    for (int c = tid; c < Dn; c += 256) {
        float a = bias[c];
#pragma unroll
        for (int f = 0; f < Fn; ++f) a += xr[f] * W[f * Dn + c];
        rowv[c] = a;
    }
    __syncthreads();
    float sm = 0.f;
    for (int i = tid; i < Dn; i += 256) sm += rowv[i];
    red[tid] = sm; __syncthreads();
    for (int st = 128; st > 0; st >>= 1) { if (tid < st) red[tid] += red[tid + st]; __syncthreads(); }
    float mean = red[0] / Dn; __syncthreads();
    float vs = 0.f;
    for (int i = tid; i < Dn; i += 256) { float d = rowv[i] - mean; vs += d * d; }
    red[tid] = vs; __syncthreads();
    for (int st = 128; st > 0; st >>= 1) { if (tid < st) red[tid] += red[tid + st]; __syncthreads(); }
    float inv = rsqrtf(red[0] / Dn + 1e-5f);
    for (int c = tid; c < Dn; c += 256) {
        float val = (rowv[c] - mean) * inv * g[c] + be[c];
        int i2 = (c >> 1) * 2;
        float ang = (float)s * __expf(-(float)i2 * (9.210340371976184f / (float)Dn));
        val += (c & 1) ? __cosf(ang) : __sinf(ang);
        h[(size_t)row * Dn + c] = val;
    }
}

// ---------------- layernorm over last dim ----------------
__global__ __launch_bounds__(256) void layernorm_k(const float* __restrict__ x,
                                                   const float* __restrict__ g,
                                                   const float* __restrict__ be,
                                                   float* __restrict__ o, int Dd) {
    __shared__ float red[256];
    int row = blockIdx.x, tid = threadIdx.x;
    const float* xr = x + (size_t)row * Dd;
    float sm = 0.f;
    for (int i = tid; i < Dd; i += 256) sm += xr[i];
    red[tid] = sm; __syncthreads();
    for (int st = 128; st > 0; st >>= 1) { if (tid < st) red[tid] += red[tid + st]; __syncthreads(); }
    float mean = red[0] / Dd; __syncthreads();
    float vs = 0.f;
    for (int i = tid; i < Dd; i += 256) { float d = xr[i] - mean; vs += d * d; }
    red[tid] = vs; __syncthreads();
    for (int st = 128; st > 0; st >>= 1) { if (tid < st) red[tid] += red[tid + st]; __syncthreads(); }
    float inv = rsqrtf(red[0] / Dd + 1e-5f);
    for (int i = tid; i < Dd; i += 256) o[(size_t)row * Dd + i] = (xr[i] - mean) * inv * g[i] + be[i];
}

// ---------------- generic WMMA GEMM: C = act(A@W + bias) (+res) ----------------
// block tile 128x128, 8 waves (4x2), wave tile 32x64 (2x4 frags), K-step 32,
// double-buffered LDS ping-pong + prefetch of tile kt+2 (global_prefetch_b8).
__global__ __launch_bounds__(256) void gemm_wmma_k(const float* __restrict__ A,
                                                   const float* __restrict__ W,
                                                   const float* __restrict__ bias,
                                                   const float* __restrict__ res,
                                                   float* __restrict__ C,
                                                   int M, int K, int N, int act) {
    __shared__ __align__(16) _Float16 As[2][128][40];   // row-major (m x k)
    __shared__ __align__(16) _Float16 BsT[2][128][40];  // transposed (n x k)
    int tid = threadIdx.x, lane = tid & 31, wave = tid >> 5;
    int wm = wave >> 1, wn = wave & 1;                  // wm 0..3, wn 0..1
    int bm = blockIdx.y * 128, bn = blockIdx.x * 128;

    CFrag acc[2][4];
#pragma unroll
    for (int mt = 0; mt < 2; ++mt)
#pragma unroll
        for (int nt = 0; nt < 4; ++nt)
#pragma unroll
            for (int r = 0; r < 8; ++r) acc[mt][nt].e[r] = 0.f;

    float4 fa[4], fb[4];
    // ---- prologue: load + store tile 0 ----
#pragma unroll
    for (int i = 0; i < 4; ++i) {
        int idx = tid + i * 256;
        fa[i] = *(const float4*)&A[(size_t)(bm + (idx >> 3)) * K + ((idx & 7) * 4)];
        fb[i] = *(const float4*)&W[(size_t)(idx >> 5) * N + bn + ((idx & 31) * 4)];
    }
#pragma unroll
    for (int i = 0; i < 4; ++i) {
        int idx = tid + i * 256;
        *(v4h*)&As[0][idx >> 3][(idx & 7) * 4] = cvt4(fa[i]);
        int r = idx >> 5, c4 = (idx & 31) * 4;
        BsT[0][c4 + 0][r] = (_Float16)fb[i].x;
        BsT[0][c4 + 1][r] = (_Float16)fb[i].y;
        BsT[0][c4 + 2][r] = (_Float16)fb[i].z;
        BsT[0][c4 + 3][r] = (_Float16)fb[i].w;
    }

    int nk = K >> 5;
    for (int kt = 0; kt < nk; ++kt) {
        __syncthreads();                       // buf[kt&1] ready for all waves
        int cur = kt & 1, nxt = cur ^ 1;
        if (kt + 1 < nk) {
            int k0 = (kt + 1) * 32;
#pragma unroll
            for (int i = 0; i < 4; ++i) {
                int idx = tid + i * 256;
                fa[i] = *(const float4*)&A[(size_t)(bm + (idx >> 3)) * K + k0 + ((idx & 7) * 4)];
                fb[i] = *(const float4*)&W[(size_t)(k0 + (idx >> 5)) * N + bn + ((idx & 31) * 4)];
            }
        }
        if (kt + 2 < nk) {
            int kp = (kt + 2) * 32;
            // gfx1250 global_prefetch_b8: warm L2/L0 for the tile after next
            __builtin_prefetch(&A[(size_t)(bm + (tid >> 1)) * K + kp], 0, 0);
            __builtin_prefetch(&W[(size_t)(kp + (tid >> 3)) * N + bn + ((tid & 7) * 16)], 0, 0);
        }
        v16h f0 = load_row_frag(&As[cur][wm * 32][0], 40, lane);
        v16h f1 = load_row_frag(&As[cur][wm * 32 + 16][0], 40, lane);
#pragma unroll
        for (int nt = 0; nt < 4; ++nt) {
            v16h bfr = load_colT_frag(&BsT[cur][wn * 64 + nt * 16][0], 40, lane);
            acc[0][nt].v = WMMA_F16(f0, bfr, acc[0][nt].v);
            acc[1][nt].v = WMMA_F16(f1, bfr, acc[1][nt].v);
        }
        if (kt + 1 < nk) {
#pragma unroll
            for (int i = 0; i < 4; ++i) {
                int idx = tid + i * 256;
                *(v4h*)&As[nxt][idx >> 3][(idx & 7) * 4] = cvt4(fa[i]);
                int r = idx >> 5, c4 = (idx & 31) * 4;
                BsT[nxt][c4 + 0][r] = (_Float16)fb[i].x;
                BsT[nxt][c4 + 1][r] = (_Float16)fb[i].y;
                BsT[nxt][c4 + 2][r] = (_Float16)fb[i].z;
                BsT[nxt][c4 + 3][r] = (_Float16)fb[i].w;
            }
        }
    }

    int rl = (lane >> 4) * 8, cl = lane & 15;
#pragma unroll
    for (int mt = 0; mt < 2; ++mt) {
#pragma unroll
        for (int nt = 0; nt < 4; ++nt) {
#pragma unroll
            for (int r = 0; r < 8; ++r) {
                int row = bm + wm * 32 + mt * 16 + rl + r;
                int col = bn + wn * 64 + nt * 16 + cl;
                float v = acc[mt][nt].e[r] + bias[col];
                if (act == 1) v = gelu_f(v);
                if (res) v += res[(size_t)row * N + col];
                C[(size_t)row * N + col] = v;
            }
        }
    }
}

// ---------------- relative position bias: relb[b,h,q,j] = q . rel[j] ----------------
__global__ __launch_bounds__(256) void relbias_k(const float* __restrict__ q,
                                                 const float* __restrict__ rel,
                                                 float* __restrict__ relb) {
    __shared__ float qrow[Dn];
    int row = blockIdx.x, tid = threadIdx.x;
    int b = row / Sn, s = row % Sn;
    for (int i = tid; i < Dn; i += 256) qrow[i] = q[(size_t)row * Dn + i];
    __syncthreads();
    for (int p = tid; p < Hn * RELN; p += 256) {
        int h = p / RELN, j = p % RELN;
        float a = 0.f;
#pragma unroll
        for (int d = 0; d < HD; ++d) a += qrow[h * HD + d] * rel[j * HD + d];
        relb[((size_t)(b * Hn + h) * Sn + s) * RELN + j] = a;
    }
}

// ---------------- flash attention with optional relative bias ----------------
// grid: x = S/128 (q block of 128 rows, 16 per wave), y = B*H
__global__ __launch_bounds__(256) void flash_attn_k(const float* __restrict__ q,
                                                    const float* __restrict__ k,
                                                    const float* __restrict__ v,
                                                    const float* __restrict__ relb,
                                                    float* __restrict__ out) {
    __shared__ __align__(16) _Float16 Kb[32][72];      // key x d (row-major)
    __shared__ __align__(16) _Float16 VbT[64][40];     // d x key (transposed)
    __shared__ __align__(16) _Float16 Qb[8][16][72];   // per-wave q tile
    __shared__ __align__(16) _Float16 Pb[8][16][40];   // per-wave p tile
    int tid = threadIdx.x, lane = tid & 31, wave = tid >> 5;
    int bh = blockIdx.y;
    int b = bh / Hn, h = bh % Hn;
    int q0 = blockIdx.x * 128 + wave * 16;

    const float* qbase = q + ((size_t)b * Sn) * Dn + h * HD;
    {
        float4 fq[8];
#pragma unroll
        for (int i = 0; i < 8; ++i) {
            int idx = lane + i * 32;
            fq[i] = *(const float4*)&qbase[(size_t)(q0 + (idx >> 4)) * Dn + ((idx & 15) * 4)];
        }
#pragma unroll
        for (int i = 0; i < 8; ++i) {
            int idx = lane + i * 32;
            *(v4h*)&Qb[wave][idx >> 4][(idx & 15) * 4] = cvt4(fq[i]);
        }
    }
    __syncthreads();
    v16h aq0 = load_row_frag(&Qb[wave][0][0], 72, lane);
    v16h aq1 = load_row_frag(&Qb[wave][0][32], 72, lane);

    float mrun[8], lrun[8];
    CFrag acc[4];
#pragma unroll
    for (int r = 0; r < 8; ++r) { mrun[r] = -1e30f; lrun[r] = 0.f; }
#pragma unroll
    for (int t = 0; t < 4; ++t)
#pragma unroll
        for (int r = 0; r < 8; ++r) acc[t].e[r] = 0.f;

    int rl = (lane >> 4) * 8, cl = lane & 15;
    int qg = q0 + rl;

    for (int kc = 0; kc < Sn; kc += 32) {
        __syncthreads();
        const float* kbp = k + ((size_t)b * Sn + kc) * Dn + h * HD;
        const float* vbp = v + ((size_t)b * Sn + kc) * Dn + h * HD;
        float4 fk[2], fv[2];
#pragma unroll
        for (int i = 0; i < 2; ++i) {
            int idx = tid + i * 256;
            int r = idx >> 4, c4 = (idx & 15) * 4;
            fk[i] = *(const float4*)&kbp[(size_t)r * Dn + c4];
            fv[i] = *(const float4*)&vbp[(size_t)r * Dn + c4];
        }
#pragma unroll
        for (int i = 0; i < 2; ++i) {
            int idx = tid + i * 256;
            int r = idx >> 4, c4 = (idx & 15) * 4;
            *(v4h*)&Kb[r][c4] = cvt4(fk[i]);
            VbT[c4 + 0][r] = (_Float16)fv[i].x;
            VbT[c4 + 1][r] = (_Float16)fv[i].y;
            VbT[c4 + 2][r] = (_Float16)fv[i].z;
            VbT[c4 + 3][r] = (_Float16)fv[i].w;
        }
        __syncthreads();

        CFrag s0, s1;
#pragma unroll
        for (int r = 0; r < 8; ++r) { s0.e[r] = 0.f; s1.e[r] = 0.f; }
        s0.v = WMMA_F16(aq0, load_colT_frag(&Kb[0][0], 72, lane), s0.v);
        s0.v = WMMA_F16(aq1, load_colT_frag(&Kb[0][32], 72, lane), s0.v);
        s1.v = WMMA_F16(aq0, load_colT_frag(&Kb[16][0], 72, lane), s1.v);
        s1.v = WMMA_F16(aq1, load_colT_frag(&Kb[16][32], 72, lane), s1.v);

        float t0[8], t1[8];
#pragma unroll
        for (int r = 0; r < 8; ++r) {
            int rowg = qg + r;
            float b0 = 0.f, b1 = 0.f;
            if (relb) {
                int c0 = kc + cl, c1 = kc + 16 + cl;
                int d0 = min(max(c0 - rowg, -Mrel), Mrel) + Mrel;
                int d1 = min(max(c1 - rowg, -Mrel), Mrel) + Mrel;
                const float* rb = relb + ((size_t)bh * Sn + rowg) * RELN;
                b0 = rb[d0]; b1 = rb[d1];
            }
            t0[r] = s0.e[r] * 0.125f + b0;
            t1[r] = s1.e[r] * 0.125f + b1;
        }
#pragma unroll
        for (int r = 0; r < 8; ++r) {
            float rm = fmaxf(t0[r], t1[r]);
#pragma unroll
            for (int off = 1; off < 16; off <<= 1) rm = fmaxf(rm, __shfl_xor(rm, off, 32));
            float mnew = fmaxf(mrun[r], rm);
            float alpha = __expf(mrun[r] - mnew);
            float p0 = __expf(t0[r] - mnew);
            float p1 = __expf(t1[r] - mnew);
            float sum = p0 + p1;
#pragma unroll
            for (int off = 1; off < 16; off <<= 1) sum += __shfl_xor(sum, off, 32);
            lrun[r] = lrun[r] * alpha + sum;
            mrun[r] = mnew;
#pragma unroll
            for (int t = 0; t < 4; ++t) acc[t].e[r] *= alpha;
            Pb[wave][rl + r][cl] = (_Float16)p0;
            Pb[wave][rl + r][16 + cl] = (_Float16)p1;
        }
        __syncthreads();
        v16h ap = load_row_frag(&Pb[wave][0][0], 40, lane);
#pragma unroll
        for (int t = 0; t < 4; ++t)
            acc[t].v = WMMA_F16(ap, load_colT_frag(&VbT[t * 16][0], 40, lane), acc[t].v);
    }

#pragma unroll
    for (int t = 0; t < 4; ++t)
#pragma unroll
        for (int r = 0; r < 8; ++r) {
            int row = q0 + rl + r;
            int col = h * HD + t * 16 + cl;
            out[((size_t)b * Sn + row) * Dn + col] = acc[t].e[r] / lrun[r];
        }
}

// ---------------- head MLPs on ctx = go[:, -1, :] ----------------
__global__ __launch_bounds__(256) void head_k(const float* __restrict__ go,
                                              const float* __restrict__ hW1,
                                              const float* __restrict__ hb1,
                                              const float* __restrict__ hg,
                                              const float* __restrict__ hbe,
                                              const float* __restrict__ hW2,
                                              const float* __restrict__ hb2,
                                              float* __restrict__ out) {
    __shared__ float ctx[Dn];
    __shared__ float red[256];
    int blk = blockIdx.x, tid = threadIdx.x;
    int b = blk / NHn, nh = blk % NHn;
    for (int i = tid; i < Dn; i += 256) ctx[i] = go[((size_t)b * Sn + (Sn - 1)) * Dn + i];
    __syncthreads();
    float y = hb1[nh * D2n + tid];
    for (int d = 0; d < Dn; ++d) y += ctx[d] * hW1[((size_t)nh * Dn + d) * D2n + tid];
    red[tid] = y; __syncthreads();
    for (int st = 128; st > 0; st >>= 1) { if (tid < st) red[tid] += red[tid + st]; __syncthreads(); }
    float mean = red[0] / D2n; __syncthreads();
    float dd = y - mean;
    red[tid] = dd * dd; __syncthreads();
    for (int st = 128; st > 0; st >>= 1) { if (tid < st) red[tid] += red[tid + st]; __syncthreads(); }
    float inv = rsqrtf(red[0] / D2n + 1e-5f); __syncthreads();
    float yn = dd * inv * hg[nh * D2n + tid] + hbe[nh * D2n + tid];
    yn = fmaxf(yn, 0.f);
    red[tid] = yn * hW2[nh * D2n + tid]; __syncthreads();
    for (int st = 128; st > 0; st >>= 1) { if (tid < st) red[tid] += red[tid + st]; __syncthreads(); }
    if (tid == 0) out[b * NHn + nh] = red[0] + hb2[nh];
}

// ---------------- orchestration ----------------
extern "C" void kernel_launch(void* const* d_in, const int* in_sizes, int n_in,
                              void* d_out, int out_size, void* d_ws, size_t ws_size,
                              hipStream_t stream) {
    (void)in_sizes; (void)n_in; (void)out_size; (void)ws_size;
    const float* X     = (const float*)d_in[0];
    const float* inW   = (const float*)d_in[1];
    const float* inb   = (const float*)d_in[2];
    const float* inlng = (const float*)d_in[3];
    const float* inlnb = (const float*)d_in[4];
    const float* ln1g  = (const float*)d_in[5];
    const float* ln1b  = (const float*)d_in[6];
    const float* Wq    = (const float*)d_in[7];
    const float* bq    = (const float*)d_in[8];
    const float* Wk    = (const float*)d_in[9];
    const float* bk    = (const float*)d_in[10];
    const float* Wv    = (const float*)d_in[11];
    const float* bv    = (const float*)d_in[12];
    const float* Wo    = (const float*)d_in[13];
    const float* bo    = (const float*)d_in[14];
    const float* rel   = (const float*)d_in[15];
    const float* ln2g  = (const float*)d_in[16];
    const float* ln2b  = (const float*)d_in[17];
    const float* W1    = (const float*)d_in[18];
    const float* b1    = (const float*)d_in[19];
    const float* W2    = (const float*)d_in[20];
    const float* b2    = (const float*)d_in[21];
    const float* gWq   = (const float*)d_in[22];
    const float* gbq   = (const float*)d_in[23];
    const float* gWk   = (const float*)d_in[24];
    const float* gbk   = (const float*)d_in[25];
    const float* gWv   = (const float*)d_in[26];
    const float* gbv   = (const float*)d_in[27];
    const float* gWo   = (const float*)d_in[28];
    const float* gbo   = (const float*)d_in[29];
    const float* hW1   = (const float*)d_in[30];
    const float* hb1   = (const float*)d_in[31];
    const float* hlng  = (const float*)d_in[32];
    const float* hlnb  = (const float*)d_in[33];
    const float* hW2   = (const float*)d_in[34];
    const float* hb2   = (const float*)d_in[35];

    const size_t ROWD = (size_t)NROW * Dn;        // 2,097,152
    float* ws    = (float*)d_ws;
    float* h     = ws;
    float* xn    = ws + ROWD;
    float* attno = ws + 2 * ROWD;
    float* qb    = ws + 3 * ROWD;
    float* kb    = qb + ROWD;
    float* vb    = kb + ROWD;
    float* relb  = vb + ROWD;                     // B*H*S*65
    float* ff1   = qb;                            // aliases q/k/v/relb region (dead during FFN)

    dim3 blk(256);
    dim3 gN512(Dn / 128, NROW / 128);             // (4, 32)
    dim3 gFF(FFn / 128, NROW / 128);              // (16, 32)
    dim3 gFlash(Sn / 128, Bn * Hn);               // (8, 32)

    input_proj_k<<<NROW, blk, 0, stream>>>(X, inW, inb, inlng, inlnb, h);

    for (int l = 0; l < Ln; ++l) {
        const float* Wq_l = Wq + (size_t)l * Dn * Dn;
        const float* Wk_l = Wk + (size_t)l * Dn * Dn;
        const float* Wv_l = Wv + (size_t)l * Dn * Dn;
        const float* Wo_l = Wo + (size_t)l * Dn * Dn;
        const float* W1_l = W1 + (size_t)l * Dn * FFn;
        const float* W2_l = W2 + (size_t)l * FFn * Dn;

        layernorm_k<<<NROW, blk, 0, stream>>>(h, ln1g + l * Dn, ln1b + l * Dn, xn, Dn);
        gemm_wmma_k<<<gN512, blk, 0, stream>>>(xn, Wq_l, bq + l * Dn, nullptr, qb, NROW, Dn, Dn, 0);
        gemm_wmma_k<<<gN512, blk, 0, stream>>>(xn, Wk_l, bk + l * Dn, nullptr, kb, NROW, Dn, Dn, 0);
        gemm_wmma_k<<<gN512, blk, 0, stream>>>(xn, Wv_l, bv + l * Dn, nullptr, vb, NROW, Dn, Dn, 0);
        relbias_k<<<NROW, blk, 0, stream>>>(qb, rel + (size_t)l * RELN * HD, relb);
        flash_attn_k<<<gFlash, blk, 0, stream>>>(qb, kb, vb, relb, attno);
        gemm_wmma_k<<<gN512, blk, 0, stream>>>(attno, Wo_l, bo + l * Dn, h, h, NROW, Dn, Dn, 0);
        layernorm_k<<<NROW, blk, 0, stream>>>(h, ln2g + l * Dn, ln2b + l * Dn, xn, Dn);
        gemm_wmma_k<<<gFF, blk, 0, stream>>>(xn, W1_l, b1 + l * FFn, nullptr, ff1, NROW, Dn, FFn, 1);
        gemm_wmma_k<<<gN512, blk, 0, stream>>>(ff1, W2_l, b2 + l * Dn, h, h, NROW, FFn, Dn, 0);
    }

    // global attention (no LN, no rel bias)
    gemm_wmma_k<<<gN512, blk, 0, stream>>>(h, gWq, gbq, nullptr, qb, NROW, Dn, Dn, 0);
    gemm_wmma_k<<<gN512, blk, 0, stream>>>(h, gWk, gbk, nullptr, kb, NROW, Dn, Dn, 0);
    gemm_wmma_k<<<gN512, blk, 0, stream>>>(h, gWv, gbv, nullptr, vb, NROW, Dn, Dn, 0);
    flash_attn_k<<<gFlash, blk, 0, stream>>>(qb, kb, vb, nullptr, attno);
    gemm_wmma_k<<<gN512, blk, 0, stream>>>(attno, gWo, gbo, nullptr, xn, NROW, Dn, Dn, 0);

    head_k<<<Bn * NHn, blk, 0, stream>>>(xn, hW1, hb1, hlng, hlnb, hW2, hb2, (float*)d_out);
}
